// MistralAttention_offloading_7584912245147
// MI455X (gfx1250) — matrix operand order repaired and checked by reading, not compiled, not run on hardware
//
#include <hip/hip_runtime.h>
#include <hip/hip_bf16.h>
#include <math.h>

typedef __bf16 bf16_t;
typedef __attribute__((ext_vector_type(16))) __bf16 v16bf;
typedef __attribute__((ext_vector_type(8)))  float  v8f;

#define NEGF  (-3.4028234663852886e38f)
#define SCALE_ATTN 0.08838834764831845f   // 1/sqrt(128)
#define LN_ROPE_THETA 9.210340371976184f  // ln(10000)

// ---------------------------------------------------------------------------
// WMMA fragment loaders (bf16, 16x16x32). Tiles staged in LDS.
// A (16xK slice, row-major [m][k]): lane<16 holds rows m=lane,
// K elements 0..15; lane>=16 holds m=lane-16, K 16..31 (same order as B).
// B (Kx16) staged TRANSPOSED in LDS as Bt[n][k] so each lane's 16 K-values are
// a contiguous 32B run: lane<16 -> n=lane, K 0..15; lane>=16 -> K 16..31.
// ---------------------------------------------------------------------------
__device__ __forceinline__ v16bf load_a_frag(const bf16_t* base, int row_stride, int lane) {
  const int m   = lane & 15;
  const int klo = (lane < 16) ? 0 : 16;
  const bf16_t* p = base + m * row_stride;
  v16bf f;
#pragma unroll
  for (int i = 0; i < 8; ++i) f[i]     = p[klo + i];
#pragma unroll
  for (int i = 0; i < 8; ++i) f[8 + i] = p[klo + 8 + i];
  return f;
}

__device__ __forceinline__ v16bf load_b_frag(const bf16_t* base, int row_stride, int lane) {
  const int n  = lane & 15;
  const int kb = (lane < 16) ? 0 : 16;
  const bf16_t* p = base + n * row_stride + kb;
  v16bf f;
#pragma unroll
  for (int i = 0; i < 16; ++i) f[i] = p[i];
  return f;
}

__device__ __forceinline__ v8f vzero8() {
  v8f s;
#pragma unroll
  for (int e = 0; e < 8; ++e) s[e] = 0.0f;
  return s;
}

__device__ __forceinline__ v8f wmma_bf16(v16bf a, v16bf b, v8f c) {
  return __builtin_amdgcn_wmma_f32_16x16x32_bf16(false, a, false, b, (short)0, c,
                                                 false, false);
}

// Load 8 consecutive f32 (two b128 loads), convert, store 8 bf16 as one b128.
__device__ __forceinline__ void cvt8_store(bf16_t* dst, const float* __restrict__ src) {
  float4 f0 = *(const float4*)(src);
  float4 f1 = *(const float4*)(src + 4);
  union { bf16_t h[8]; uint4 u; } pk;
  pk.h[0] = (bf16_t)f0.x; pk.h[1] = (bf16_t)f0.y;
  pk.h[2] = (bf16_t)f0.z; pk.h[3] = (bf16_t)f0.w;
  pk.h[4] = (bf16_t)f1.x; pk.h[5] = (bf16_t)f1.y;
  pk.h[6] = (bf16_t)f1.z; pk.h[7] = (bf16_t)f1.w;
  *(uint4*)dst = pk.u;
}

// ---------------------------------------------------------------------------
// GEMM: C[M,N] (f32) = A[M,K] (f32) x B[K,N] (f32), bf16 WMMA inner product.
// Templated on N,K so all strides are compile-time -> immediate-offset loads.
// 256 threads = 8 waves; block tile 64x128; wave tile 32x32 (2x2 frags).
// Double-buffered LDS: stage k+1 while WMMAs consume k.
// ---------------------------------------------------------------------------
#define GM_TM 64
#define GM_TN 128
#define GM_TK 32

template <int N, int K>
__global__ __launch_bounds__(256) void gemm_f32_bf16wmma(
    const float* __restrict__ A, const float* __restrict__ B,
    float* __restrict__ C) {
  __shared__ __align__(16) bf16_t As[2][GM_TM][GM_TK];   // [m][k]
  __shared__ __align__(16) bf16_t Bs[2][GM_TN][GM_TK];   // transposed: [n][k]

  const int tid  = threadIdx.x;
  const int lane = tid & 31;
  const int wave = tid >> 5;
  const int m0   = blockIdx.x * GM_TM;
  const int n0   = blockIdx.y * GM_TN;
  const int mw   = (wave & 1) * 32;
  const int nw   = (wave >> 1) * 32;

  // staging coordinates (exact work per thread, no dynamic loops)
  const int ar  = tid >> 2;             // A: row 0..63
  const int ac  = (tid & 3) * 8;        // A: 8 consecutive k
  const int bn  = tid & 127;            // B: fixed n
  const int bkh = (tid >> 7) * 16;      // B: 16-k half

  const float* aptr = A + (size_t)(m0 + ar) * K + ac;   // + k0 per step
  const float* bptr = B + (size_t)bkh * N + n0 + bn;    // + k0*N per step

  v8f acc[2][2];
#pragma unroll
  for (int i = 0; i < 2; ++i)
#pragma unroll
    for (int j = 0; j < 2; ++j) acc[i][j] = vzero8();

  auto stage = [&](int buf, int k0) {
    cvt8_store(&As[buf][ar][ac], aptr + k0);
    const float* bp = bptr + (size_t)k0 * N;
    union { bf16_t h[16]; uint4 u[2]; } pk;
#pragma unroll
    for (int kk = 0; kk < 16; ++kk)
      pk.h[kk] = (bf16_t)bp[(size_t)kk * N];            // constant offsets
    uint4* bdst = (uint4*)&Bs[buf][bn][bkh];
    bdst[0] = pk.u[0];
    bdst[1] = pk.u[1];
  };

  stage(0, 0);
  __syncthreads();

  for (int k0 = 0; k0 < K; k0 += GM_TK) {
    const int cur = (k0 >> 5) & 1;
    if (k0 + GM_TK < K) stage(cur ^ 1, k0 + GM_TK);

    v16bf bfr[2];
#pragma unroll
    for (int j = 0; j < 2; ++j)
      bfr[j] = load_b_frag(&Bs[cur][nw + 16 * j][0], GM_TK, lane);
#pragma unroll
    for (int i = 0; i < 2; ++i) {
      v16bf afr = load_a_frag(&As[cur][mw + 16 * i][0], GM_TK, lane);
#pragma unroll
      for (int j = 0; j < 2; ++j) acc[i][j] = wmma_bf16(afr, bfr[j], acc[i][j]);
    }
    __syncthreads();
  }

  // C fragment layout: VGPR e holds M = 16*i + e (+8 for lanes>=16), N = lane%16
#pragma unroll
  for (int i = 0; i < 2; ++i)
#pragma unroll
    for (int j = 0; j < 2; ++j) {
      int mbase = m0 + mw + 16 * i + ((lane < 16) ? 0 : 8);
      int n     = n0 + nw + 16 * j + (lane & 15);
#pragma unroll
      for (int e = 0; e < 8; ++e)
        C[(size_t)(mbase + e) * N + n] = acc[i][j][e];
    }
}

// ---------------------------------------------------------------------------
// RoPE in-place on q [1024, 32*128] and k [1024, 8*128].
// grid (1024 pos, 40 heads), block 64 (one thread per rotation pair).
// ---------------------------------------------------------------------------
__global__ void rope_kernel(float* __restrict__ q, float* __restrict__ k) {
  const int pos  = blockIdx.x;
  const int head = blockIdx.y;  // 0..31 = q heads, 32..39 = k heads
  const int d    = threadIdx.x; // 0..63
  float inv = __expf(-(2.0f * (float)d / 128.0f) * LN_ROPE_THETA);
  float ang = (float)pos * inv;
  float c = __cosf(ang), s = __sinf(ang);
  float* p = (head < 32) ? (q + (size_t)pos * 4096 + head * 128)
                         : (k + (size_t)pos * 1024 + (head - 32) * 128);
  float x1 = p[d], x2 = p[d + 64];
  p[d]      = x1 * c - x2 * s;
  p[d + 64] = x2 * c + x1 * s;
}

// ---------------------------------------------------------------------------
// Label build: gather 16 outlier channels by sorted_channel[h], 4-bit pseudo-
// quantize per 16-vector. One wave per (pos,h): lanes 0-15 -> gq, 16-31 -> gk.
// ---------------------------------------------------------------------------
__global__ void label_kernel(const float* __restrict__ q, const float* __restrict__ k,
                             const int* __restrict__ sc,
                             float* __restrict__ gq, float* __restrict__ gk) {
  const int pos = blockIdx.x;
  const int h   = blockIdx.y;
  const int t   = threadIdx.x;      // 0..31
  const int j   = t & 15;
  const bool isK = (t >= 16);
  const int ch = sc[h * 128 + j];
  float val = isK ? k[(size_t)pos * 1024 + (h >> 2) * 128 + ch]
                  : q[(size_t)pos * 4096 + h * 128 + ch];
  float mn = val, mx = val;
#pragma unroll
  for (int off = 8; off >= 1; off >>= 1) {
    mn = fminf(mn, __shfl_xor(mn, off, 16));
    mx = fmaxf(mx, __shfl_xor(mx, off, 16));
  }
  float rng = mx - mn;
  if (rng == 0.0f) rng = 1.0f;
  float scale = 15.0f / rng;
  float qv = fminf(fmaxf(rintf((val - mn) * scale), 0.0f), 15.0f);
  float dq = qv / scale + mn;
  float* dst = isK ? gk : gq;
  dst[((size_t)h * 1024 + pos) * 16 + j] = dq;
}

// ---------------------------------------------------------------------------
// Top-256 radix select per (h, qpos) over 1024 label scores.
// Scores mapped to order-preserving uint32; 32-bit MSB-first radix descent.
// Emits keep-mask: 32 x uint32 per row.
// ---------------------------------------------------------------------------
__global__ __launch_bounds__(128) void select_kernel(
    const float* __restrict__ gq, const float* __restrict__ gk,
    unsigned* __restrict__ keep) {
  __shared__ unsigned um[1024];
  __shared__ int cnt;
  const int qpos = blockIdx.x;
  const int h    = blockIdx.y;
  const int tid  = threadIdx.x;

  const float4* gq4 = (const float4*)(gq + ((size_t)h * 1024 + qpos) * 16);
  float4 qv[4];
#pragma unroll
  for (int i = 0; i < 4; ++i) qv[i] = gq4[i];

  unsigned mloc[8];
#pragma unroll
  for (int i = 0; i < 8; ++i) {
    int key = tid * 8 + i;
    const float4* gk4 = (const float4*)(gk + ((size_t)h * 1024 + key) * 16);
    float s = 0.0f;
#pragma unroll
    for (int j = 0; j < 4; ++j) {
      float4 kv = gk4[j];
      s += qv[j].x * kv.x + qv[j].y * kv.y + qv[j].z * kv.z + qv[j].w * kv.w;
    }
    s *= 0.25f;                      // 1/sqrt(16)
    if (key > qpos) s += NEGF;       // causal mask (finfo.min like reference)
    unsigned u = __float_as_uint(s);
    u = (u & 0x80000000u) ? ~u : (u | 0x80000000u);
    mloc[i] = u;
    um[key] = u;
  }

  unsigned prefix = 0;
  int need = 256;
  for (int bit = 31; bit >= 0; --bit) {
    if (tid == 0) cnt = 0;
    __syncthreads();
    unsigned test   = prefix | (1u << bit);
    unsigned himask = ~((1u << bit) - 1u);
    int c = 0;
#pragma unroll
    for (int i = 0; i < 8; ++i) c += ((mloc[i] & himask) == test) ? 1 : 0;
    atomicAdd(&cnt, c);
    __syncthreads();
    int total = cnt;
    if (total >= need) prefix = test; else need -= total;
    __syncthreads();
  }

  if (tid < 32) {
    unsigned w = 0;
#pragma unroll 4
    for (int b = 0; b < 32; ++b)
      if (um[tid * 32 + b] >= prefix) w |= (1u << b);
    keep[((size_t)h * 1024 + qpos) * 32 + tid] = w;
  }
}

// ---------------------------------------------------------------------------
// Fused flash attention per (head, 64 q-rows). Two passes over 32-key blocks:
// pass1: S = Q K^T via WMMA -> running row max + sumexp (keep-mask + causal);
// pass2: recompute S, P = exp(S - m) (deferred 1/l), O += P V via WMMA.
// ---------------------------------------------------------------------------
#define AT_BM 64
#define AT_BN 32

__global__ __launch_bounds__(256) void attn_kernel(
    const float* __restrict__ q, const float* __restrict__ k,
    const float* __restrict__ v, const unsigned* __restrict__ keep,
    float* __restrict__ out) {
  __shared__ __align__(16) bf16_t Qt[AT_BM][128];
  __shared__ __align__(16) bf16_t Kt[AT_BN][128];   // [key][d] == Bt layout for S
  __shared__ __align__(16) bf16_t Vt[128][AT_BN];   // transposed: [d][key]
  __shared__ float  St[AT_BM][AT_BN];
  __shared__ __align__(16) bf16_t Pb[AT_BM][AT_BN];
  __shared__ float  mrow[AT_BM], lrow[AT_BM];

  const int h    = blockIdx.y;
  const int q0   = blockIdx.x * AT_BM;
  const int kvh  = h >> 2;            // GQA group = 4
  const int tid  = threadIdx.x;
  const int lane = tid & 31;
  const int wave = tid >> 5;

  // stage Q tile, f32 -> bf16 (4 x 8-element vector chunks per thread)
#pragma unroll
  for (int rep = 0; rep < 4; ++rep) {
    int idx = (rep * 256 + tid) * 8;
    int r = idx >> 7, c = idx & 127;
    cvt8_store(&Qt[r][c], &q[(size_t)(q0 + r) * 4096 + h * 128 + c]);
  }
  if (tid < AT_BM) { mrow[tid] = -3.0e38f; lrow[tid] = 0.0f; }
  __syncthreads();

  const int nb  = q0 / AT_BN + 2;     // key blocks covering keys <= q0+63
  const int smw = wave & 3;           // S: 4 m-groups x 2 n-groups
  const int snw = wave >> 2;
  const int vd  = tid & 127;          // V staging: fixed d
  const int vkh = (tid >> 7) * 16;    // V staging: 16-key half
  const float* vbase = v + (size_t)vkh * 1024 + kvh * 128 + vd;

  // ---------------- pass 1: row max / sumexp ----------------
  for (int kb = 0; kb < nb; ++kb) {
    const int kbase = kb * AT_BN;
#pragma unroll
    for (int rep = 0; rep < 2; ++rep) {
      int idx = (rep * 256 + tid) * 8;
      int r = idx >> 7, c = idx & 127;
      cvt8_store(&Kt[r][c], &k[(size_t)(kbase + r) * 1024 + kvh * 128 + c]);
    }
    __syncthreads();

    v8f s = vzero8();
#pragma unroll
    for (int kk = 0; kk < 128; kk += 32) {
      v16bf af = load_a_frag(&Qt[smw * 16][kk], 128, lane);
      v16bf bf = load_b_frag(&Kt[snw * 16][kk], 128, lane);
      s = wmma_bf16(af, bf, s);
    }
    {
      int mbase = smw * 16 + ((lane < 16) ? 0 : 8);
      int n     = snw * 16 + (lane & 15);
#pragma unroll
      for (int e = 0; e < 8; ++e) St[mbase + e][n] = s[e];
    }
    __syncthreads();

    if (tid < AT_BM) {
      const int qg = q0 + tid;
      const unsigned kw = keep[((size_t)h * 1024 + qg) * 32 + (kbase >> 5)];
      float mold = mrow[tid];
      float mmax = mold;
#pragma unroll 8
      for (int n = 0; n < AT_BN; ++n) {
        bool kept = ((kbase + n) <= qg) && ((kw >> n) & 1u);
        float sv = kept ? St[tid][n] * SCALE_ATTN : -__builtin_inff();
        mmax = fmaxf(mmax, sv);
      }
      float l = lrow[tid] * __expf(mold - mmax);
#pragma unroll 8
      for (int n = 0; n < AT_BN; ++n) {
        bool kept = ((kbase + n) <= qg) && ((kw >> n) & 1u);
        float sv = kept ? St[tid][n] * SCALE_ATTN : -__builtin_inff();
        l += __expf(sv - mmax);
      }
      mrow[tid] = mmax; lrow[tid] = l;
    }
    __syncthreads();
  }

  // ---------------- pass 2: O accumulation ----------------
  const int omw = wave & 3;           // O: 4 m-groups x 2 n-halves (64 cols)
  const int onw = wave >> 2;
  v8f oacc[4];
#pragma unroll
  for (int j = 0; j < 4; ++j) oacc[j] = vzero8();

  for (int kb = 0; kb < nb; ++kb) {
    const int kbase = kb * AT_BN;
#pragma unroll
    for (int rep = 0; rep < 2; ++rep) {
      int idx = (rep * 256 + tid) * 8;
      int r = idx >> 7, c = idx & 127;
      cvt8_store(&Kt[r][c], &k[(size_t)(kbase + r) * 1024 + kvh * 128 + c]);
    }
    {   // stage V transposed: fixed d, 16-key run, constant-offset loads
      const float* vp = vbase + (size_t)kbase * 1024;
      union { bf16_t hh[16]; uint4 u[2]; } pk;
#pragma unroll
      for (int kk = 0; kk < 16; ++kk)
        pk.hh[kk] = (bf16_t)vp[(size_t)kk * 1024];    // immediate offsets
      uint4* vdst = (uint4*)&Vt[vd][vkh];
      vdst[0] = pk.u[0];
      vdst[1] = pk.u[1];
    }
    __syncthreads();

    v8f s = vzero8();
#pragma unroll
    for (int kk = 0; kk < 128; kk += 32) {
      v16bf af = load_a_frag(&Qt[smw * 16][kk], 128, lane);
      v16bf bf = load_b_frag(&Kt[snw * 16][kk], 128, lane);
      s = wmma_bf16(af, bf, s);
    }
    {
      int mbase = smw * 16 + ((lane < 16) ? 0 : 8);
      int n     = snw * 16 + (lane & 15);
#pragma unroll
      for (int e = 0; e < 8; ++e) St[mbase + e][n] = s[e];
    }
    __syncthreads();

    if (tid < AT_BM) {
      const int qg = q0 + tid;
      const unsigned kw = keep[((size_t)h * 1024 + qg) * 32 + (kbase >> 5)];
      const float mfin = mrow[tid];
#pragma unroll 8
      for (int n = 0; n < AT_BN; ++n) {
        bool kept = ((kbase + n) <= qg) && ((kw >> n) & 1u);
        float p = kept ? __expf(St[tid][n] * SCALE_ATTN - mfin) : 0.0f;
        Pb[tid][n] = (bf16_t)p;
      }
    }
    __syncthreads();

    v16bf pa = load_a_frag(&Pb[omw * 16][0], AT_BN, lane);
#pragma unroll
    for (int j = 0; j < 4; ++j) {
      v16bf vb = load_b_frag(&Vt[onw * 64 + j * 16][0], AT_BN, lane);
      oacc[j] = wmma_bf16(pa, vb, oacc[j]);
    }
    __syncthreads();
  }

  // epilogue: scale by 1/l, write [q, h*128 + d]
  {
    int mb = omw * 16 + ((lane < 16) ? 0 : 8);
#pragma unroll
    for (int j = 0; j < 4; ++j) {
      int n = onw * 64 + j * 16 + (lane & 15);
#pragma unroll
      for (int e = 0; e < 8; ++e) {
        int m = mb + e;
        float linv = 1.0f / lrow[m];
        out[(size_t)(q0 + m) * 4096 + h * 128 + n] = oacc[j][e] * linv;
      }
    }
  }
}

// ---------------------------------------------------------------------------
// Host launcher
// ---------------------------------------------------------------------------
extern "C" void kernel_launch(void* const* d_in, const int* in_sizes, int n_in,
                              void* d_out, int out_size, void* d_ws, size_t ws_size,
                              hipStream_t stream) {
  (void)in_sizes; (void)n_in; (void)out_size; (void)ws_size;
  const float* hidden = (const float*)d_in[0];
  // d_in[1] attention_mask: causal mask derived analytically
  // d_in[2] position_ids:   positions are arange(Q)
  const float* Wq = (const float*)d_in[3];
  const float* Wk = (const float*)d_in[4];
  const float* Wv = (const float*)d_in[5];
  const float* Wo = (const float*)d_in[6];
  const int*   sc = (const int*)d_in[7];
  float* out = (float*)d_out;

  float* ws   = (float*)d_ws;
  float* qb   = ws;                                   // 1024*4096
  float* kb   = qb + (size_t)1024 * 4096;             // 1024*1024
  float* vb   = kb + (size_t)1024 * 1024;             // 1024*1024
  float* gq   = vb + (size_t)1024 * 1024;             // 32*1024*16
  float* gk   = gq + (size_t)32 * 1024 * 16;          // 32*1024*16
  unsigned* keep = (unsigned*)(gk + (size_t)32 * 1024 * 16);  // 32*1024*32 words
  float* aout = (float*)(keep + (size_t)32 * 1024 * 32);      // 1024*4096

  dim3 blk(256);
  // QKV projections (bf16 WMMA), compile-time strides
  gemm_f32_bf16wmma<4096, 4096><<<dim3(1024 / GM_TM, 4096 / GM_TN), blk, 0, stream>>>(hidden, Wq, qb);
  gemm_f32_bf16wmma<1024, 4096><<<dim3(1024 / GM_TM, 1024 / GM_TN), blk, 0, stream>>>(hidden, Wk, kb);
  gemm_f32_bf16wmma<1024, 4096><<<dim3(1024 / GM_TM, 1024 / GM_TN), blk, 0, stream>>>(hidden, Wv, vb);
  // RoPE on q and k
  rope_kernel<<<dim3(1024, 40), dim3(64), 0, stream>>>(qb, kb);
  // labels (gather + 4-bit pseudo-quant)
  label_kernel<<<dim3(1024, 32), dim3(32), 0, stream>>>(qb, kb, sc, gq, gk);
  // top-256 keep masks
  select_kernel<<<dim3(1024, 32), dim3(128), 0, stream>>>(gq, gk, keep);
  // fused flash attention with keep-mask
  attn_kernel<<<dim3(1024 / AT_BM, 32), blk, 0, stream>>>(qb, kb, vb, keep, aout);
  // output projection
  gemm_f32_bf16wmma<4096, 4096><<<dim3(1024 / GM_TM, 4096 / GM_TN), blk, 0, stream>>>(aout, Wo, out);
}
